// MyRNN_33011118637832
// MI455X (gfx1250) — matrix operand order repaired and checked by reading, not compiled, not run on hardware
//
#include <hip/hip_runtime.h>
#include <hip/hip_bf16.h>

// ---------------------------------------------------------------------------
// 2-layer LSTM (B=2048, T=80, EMB=100->pad128, U=512) + Dense(1)+sigmoid.
// bf16 WMMA (v_wmma_f32_16x16x32_bf16), f32 accumulate, fused LSTM cell
// epilogue fully in registers (each wave owns all 4 gates of its units).
// Branch-free transcendentals (native v_tanh_f32 when available).
// ---------------------------------------------------------------------------

typedef __bf16 bf16;
typedef __attribute__((ext_vector_type(16))) __bf16 v16bf;
typedef __attribute__((ext_vector_type(8)))  __bf16 v8bf;
typedef __attribute__((ext_vector_type(8)))  float  v8f;

#define BATCH 2048
#define SEQ   80
#define EMB   100
#define EMBP  128     // padded K for layer-1 x part
#define HU    512     // units
#define NG    2048    // 4*HU gate columns
#define KT1   (EMBP + HU)   // 640
#define KT2   (HU + HU)     // 1024

// ---------------- fragment loads (CDNA5 wave32 WMMA layouts) ----------------
// A (16x32 bf16, row-major src, stride ld):
//   lane<16 : M=lane,    K = {k0+0..7,  k0+16..23}
//   lane>=16: M=lane-16, K = {k0+8..15, k0+24..31}
__device__ inline v16bf load_a_frag(const bf16* __restrict__ A, int ld, int row0, int k0) {
    const int lane = threadIdx.x & 31;
    const bf16* p = A + (size_t)(row0 + (lane & 15)) * ld + k0 + ((lane >> 4) << 3);
    v8bf lo = *(const v8bf*)(p);
    v8bf hi = *(const v8bf*)(p + 16);
    v16bf r;
#pragma unroll
    for (int i = 0; i < 8; ++i) { r[i] = lo[i]; r[i + 8] = hi[i]; }
    return r;
}

// B (32x16 bf16) from N-major weights Wt[n][k] (stride ld = Ktot):
//   lane<16 : N=lane,    K = {k0+0..7,  k0+16..23}
//   lane>=16: N=lane-16, K = {k0+8..15, k0+24..31}
__device__ inline v16bf load_b_frag(const bf16* __restrict__ Wt, int ld, int n0, int k0) {
    const int lane = threadIdx.x & 31;
    const bf16* p = Wt + (size_t)(n0 + (lane & 15)) * ld + k0 + ((lane >> 4) << 3);
    v8bf lo = *(const v8bf*)(p);
    v8bf hi = *(const v8bf*)(p + 16);
    v16bf r;
#pragma unroll
    for (int i = 0; i < 8; ++i) { r[i] = lo[i]; r[i + 8] = hi[i]; }
    return r;
}

__device__ inline float sigmoidf_(float x) { return 1.0f / (1.0f + __expf(-x)); }

// Branch-free tanh: native CDNA5 v_tanh_f32 if the builtin exists, else a
// divergence-free exp-based formulation (libm tanhf's branchy path splits
// EXEC and serializes the epilogue).
__device__ inline float fast_tanhf(float x) {
#if __has_builtin(__builtin_amdgcn_tanhf)
    return __builtin_amdgcn_tanhf(x);
#else
    float e = __expf(-2.0f * x);           // v_exp_f32
    return (1.0f - e) / (1.0f + e);        // exact sign/saturation behavior
#endif
}

// ---------------- prep kernels ----------------
// Xemb[t][b][0..127] = bf16(emb[ids[b][t]][k]) (zero-pad k>=100)
__global__ void embed_kernel(const int* __restrict__ ids, const float* __restrict__ emb,
                             bf16* __restrict__ X) {
    int gid = blockIdx.x * 256 + threadIdx.x;
    const int total = SEQ * BATCH * EMBP;
    if (gid >= total) return;
    int k  = gid & (EMBP - 1);
    int bt = gid >> 7;          // t*BATCH + b
    int b  = bt & (BATCH - 1);  // BATCH = 2^11
    int t  = bt >> 11;
    float v = (k < EMB) ? emb[(size_t)ids[b * SEQ + t] * EMB + k] : 0.0f;
    X[gid] = (bf16)v;
}

// Wc[n][k] (bf16, N-major): k<K1 -> W[k][n] (zero-pad k>=Kw), else U[k-K1][n]
__global__ void prep_wc_kernel(const float* __restrict__ W, const float* __restrict__ U,
                               bf16* __restrict__ Wc, int K1, int Kw, int Ktot) {
    int gid = blockIdx.x * 256 + threadIdx.x;
    int total = NG * Ktot;
    if (gid >= total) return;
    int n = gid / Ktot, k = gid - n * Ktot;
    float v;
    if (k < K1) v = (k < Kw) ? W[(size_t)k * NG + n] : 0.0f;
    else        v = U[(size_t)(k - K1) * NG + n];
    Wc[gid] = (bf16)v;
}

// ---------------- fused GEMM + LSTM cell step ----------------
// z[b][g*512+u] = sum_k concat(A1[b][:K1], Hprev[b][:512])[k] * Wc[g*512+u][k]
// then i,f,g,o gates -> c,h update. Wave = 64 rows x 16 units x 4 gates.
__global__ __launch_bounds__(256) void lstm_step_kernel(
    const bf16* __restrict__ A1, int lda1, int K1,
    const bf16* __restrict__ Hprev,
    const bf16* __restrict__ Wc, int Ktot,
    const float* __restrict__ bias,
    float* __restrict__ Cst,
    bf16* __restrict__ Hout) {

    const int lane = threadIdx.x & 31;
    const int w    = threadIdx.x >> 5;               // 8 waves
    const int u0   = blockIdx.x * 64 + (w & 3) * 16; // 16 units per wave
    const int row0 = blockIdx.y * 128 + (w >> 2) * 64;

    v8f acc[4][4]; // [gate][row-tile]
    const v8f zacc = {0.f, 0.f, 0.f, 0.f, 0.f, 0.f, 0.f, 0.f};
#pragma unroll
    for (int g = 0; g < 4; ++g)
#pragma unroll
        for (int r = 0; r < 4; ++r) acc[g][r] = zacc;

    for (int k0 = 0; k0 < Ktot; k0 += 32) {
        const bf16* src; int ld, kk;
        if (k0 < K1) { src = A1;    ld = lda1; kk = k0; }
        else         { src = Hprev; ld = HU;   kk = k0 - K1; }
        v16bf a[4], bm[4];
#pragma unroll
        for (int r = 0; r < 4; ++r) a[r] = load_a_frag(src, ld, row0 + r * 16, kk);
#pragma unroll
        for (int g = 0; g < 4; ++g) bm[g] = load_b_frag(Wc, Ktot, g * HU + u0, k0);
#pragma unroll
        for (int g = 0; g < 4; ++g)
#pragma unroll
            for (int r = 0; r < 4; ++r)
                acc[g][r] = __builtin_amdgcn_wmma_f32_16x16x32_bf16(
                    false, a[r], false, bm[g], (short)0, acc[g][r], false, false);
    }

    // Cell update: acc[0..3] are i,f,g,o for identical (row,unit) lane slots.
    const int unit = u0 + (lane & 15);
    const float bi  = bias[unit];
    const float bff = bias[HU + unit];
    const float bg  = bias[2 * HU + unit];
    const float bo  = bias[3 * HU + unit];
#pragma unroll
    for (int r = 0; r < 4; ++r) {
#pragma unroll
        for (int v = 0; v < 8; ++v) {
            int row = row0 + r * 16 + v + ((lane >> 4) << 3);
            float ig = sigmoidf_(acc[0][r][v] + bi);
            float fg = sigmoidf_(acc[1][r][v] + bff);
            float gg = fast_tanhf(acc[2][r][v] + bg);
            float og = sigmoidf_(acc[3][r][v] + bo);
            size_t idx = (size_t)row * HU + unit;
            float cn = fg * Cst[idx] + ig * gg;
            Cst[idx]  = cn;
            Hout[idx] = (bf16)(og * fast_tanhf(cn));
        }
    }
}

// ---------------- dense head ----------------
__global__ void dense_kernel(const bf16* __restrict__ h, const float* __restrict__ Wd,
                             const float* __restrict__ bd, float* __restrict__ out) {
    int b = blockIdx.x * blockDim.x + threadIdx.x;
    if (b >= BATCH) return;
    float s = 0.f;
#pragma unroll 8
    for (int k = 0; k < HU; ++k) s += (float)h[(size_t)b * HU + k] * Wd[k];
    out[b] = sigmoidf_(s + bd[0]);
}

// ---------------- host launcher ----------------
extern "C" void kernel_launch(void* const* d_in, const int* in_sizes, int n_in,
                              void* d_out, int out_size, void* d_ws, size_t ws_size,
                              hipStream_t stream) {
    (void)in_sizes; (void)n_in; (void)out_size; (void)ws_size;
    const int*   ids = (const int*)d_in[0];
    const float* emb = (const float*)d_in[1];
    const float* W1  = (const float*)d_in[2];
    const float* U1  = (const float*)d_in[3];
    const float* b1  = (const float*)d_in[4];
    const float* W2  = (const float*)d_in[5];
    const float* U2  = (const float*)d_in[6];
    const float* b2  = (const float*)d_in[7];
    const float* Wd  = (const float*)d_in[8];
    const float* bd  = (const float*)d_in[9];
    float* out = (float*)d_out;

    // workspace carve-up (256B aligned)
    char* ws = (char*)d_ws;
    size_t off = 0;
    auto carve = [&](size_t bytes) { size_t o = off; off = (off + bytes + 255) & ~(size_t)255; return o; };
    bf16*  Xemb  = (bf16*)(ws + carve((size_t)SEQ * BATCH * EMBP * sizeof(bf16)));
    bf16*  Wc1   = (bf16*)(ws + carve((size_t)NG * KT1 * sizeof(bf16)));
    bf16*  Wc2   = (bf16*)(ws + carve((size_t)NG * KT2 * sizeof(bf16)));
    bf16*  h1b[2]; h1b[0] = (bf16*)(ws + carve((size_t)BATCH * HU * sizeof(bf16)));
                   h1b[1] = (bf16*)(ws + carve((size_t)BATCH * HU * sizeof(bf16)));
    bf16*  h2b[2]; h2b[0] = (bf16*)(ws + carve((size_t)BATCH * HU * sizeof(bf16)));
                   h2b[1] = (bf16*)(ws + carve((size_t)BATCH * HU * sizeof(bf16)));
    float* c1 = (float*)(ws + carve((size_t)BATCH * HU * sizeof(float)));
    float* c2 = (float*)(ws + carve((size_t)BATCH * HU * sizeof(float)));

    // zero initial states (graph-capture-safe async memsets)
    hipMemsetAsync(h1b[0], 0, (size_t)BATCH * HU * sizeof(bf16), stream);
    hipMemsetAsync(h2b[0], 0, (size_t)BATCH * HU * sizeof(bf16), stream);
    hipMemsetAsync(c1, 0, (size_t)BATCH * HU * sizeof(float), stream);
    hipMemsetAsync(c2, 0, (size_t)BATCH * HU * sizeof(float), stream);

    // prep: embedding gather + combined transposed bf16 weights
    {
        int total = SEQ * BATCH * EMBP;
        embed_kernel<<<(total + 255) / 256, 256, 0, stream>>>(ids, emb, Xemb);
    }
    {
        int total = NG * KT1;
        prep_wc_kernel<<<(total + 255) / 256, 256, 0, stream>>>(W1, U1, Wc1, EMBP, EMB, KT1);
    }
    {
        int total = NG * KT2;
        prep_wc_kernel<<<(total + 255) / 256, 256, 0, stream>>>(W2, U2, Wc2, HU, HU, KT2);
    }

    // recurrence: 80 steps x 2 layers, h double-buffered
    dim3 grid(NG / (4 * 64) /*=8 unit blocks*/, BATCH / 128 /*=16 row blocks*/);
    for (int t = 0; t < SEQ; ++t) {
        const bf16* x1 = Xemb + (size_t)t * BATCH * EMBP;  // time-major slab, lda=EMBP
        bf16* h1n = h1b[(t + 1) & 1];
        bf16* h2n = h2b[(t + 1) & 1];
        lstm_step_kernel<<<grid, 256, 0, stream>>>(x1, EMBP, EMBP, h1b[t & 1],
                                                   Wc1, KT1, b1, c1, h1n);
        lstm_step_kernel<<<grid, 256, 0, stream>>>(h1n, HU, HU, h2b[t & 1],
                                                   Wc2, KT2, b2, c2, h2n);
    }

    // dense head on final h2 (t=79 wrote buffer 0)
    dense_kernel<<<BATCH / 256, 256, 0, stream>>>(h2b[0], Wd, bd, out);
}
